// InteractionTransformer_35021163331836
// MI455X (gfx1250) — matrix operand order repaired and checked by reading, not compiled, not run on hardware
//
#include <hip/hip_runtime.h>
#include <hip/hip_bf16.h>
#include <math.h>

// ---------------- problem constants ----------------
#define BB   8
#define NQ   300
#define CC   256
#define NHD  8
#define HD   32
#define NL   4
#define NP   4
#define DFF  1024
#define LIN  13294
#define ROWS (BB*NQ)          // 2400
#define SRCROWS (BB*LIN)      // 106352

typedef __attribute__((ext_vector_type(16))) _Float16 v16h;
typedef __attribute__((ext_vector_type(8)))  _Float16 v8h;
typedef __attribute__((ext_vector_type(4)))  _Float16 v4h;
typedef __attribute__((ext_vector_type(8)))  float    v8f;

__device__ __forceinline__ float warp_sum(float v) {
  #pragma unroll
  for (int o = 16; o > 0; o >>= 1) v += __shfl_xor(v, o, 32);
  return v;
}
__device__ __forceinline__ float warp_max(float v) {
  #pragma unroll
  for (int o = 16; o > 0; o >>= 1) v = fmaxf(v, __shfl_xor(v, o, 32));
  return v;
}

// ---------------------------------------------------------------------------
// Generic GEMM: C[M,N] = act(A[M,K] @ B[K,N] + bias)
// Requirements (all call sites satisfy): K % 32 == 0, N % 128 == 0,
// bias non-null, A/B 16-byte aligned, lda/ldb % 4 == 0.
// Block: 256 threads = 8 waves (2 wm x 4 wn); wave computes 16x32 output
// (2 accumulators sharing one A fragment). Block tile: 32 x 128, k-step 32.
// A staged row-major f16 (padded), B staged TRANSPOSED f16 (sBt[n][k]) so
// both fragments are two ds_load_b128 each.
// FLAGS (compile-time): bit0 = relu, bit1 = store output as _Float16
// ---------------------------------------------------------------------------
template <int FLAGS>
__global__ __launch_bounds__(256) void gemm_wmma_f16(
    const float* __restrict__ A, int lda,
    const float* __restrict__ B, int ldb,
    const float* __restrict__ bias,
    void* __restrict__ Cout, int ldc,
    int M, int N, int K)
{
  __shared__ _Float16 sA[32][40];    // row-major, 80B row stride (16B aligned)
  __shared__ _Float16 sBt[128][40];  // transposed: [n][k]

  const int tid  = threadIdx.x;
  const int lane = tid & 31;
  const int w    = tid >> 5;
  const int wm   = w >> 2;           // 0..1
  const int wn   = w & 3;            // 0..3
  const int lhi  = lane >> 4;        // 0/1
  const int lm   = lane & 15;
  const int m0 = blockIdx.y * 32;
  const int n0 = blockIdx.x * 128;

  // A staging coords: one float4 per thread (32 rows x 8 float4)
  const int ar  = tid >> 3;          // 0..31
  const int ac4 = (tid & 7) << 2;    // 0,4,...,28
  const int gm  = m0 + ar;
  const int gmc = gm < M ? gm : (M - 1);
  const float apm = (gm < M) ? 1.0f : 0.0f;

  v8f c0 = {}, c1 = {};

  for (int k0 = 0; k0 < K; k0 += 32) {
    // ---- stage A tile (32 x 32) : coalesced float4, branchless edge ----
    {
      const float4* ap =
          (const float4*)(A + (long long)gmc * lda + (k0 + ac4));
      float4 v = *ap;
      __builtin_prefetch(ap + 8, 0, 1);   // next k-tile (speculative)
      v4h hv;
      hv[0] = (_Float16)(v.x * apm); hv[1] = (_Float16)(v.y * apm);
      hv[2] = (_Float16)(v.z * apm); hv[3] = (_Float16)(v.w * apm);
      *(v4h*)&sA[ar][ac4] = hv;
    }
    // ---- stage B tile (32k x 128n), transposed into sBt[n][k] ----
    #pragma unroll
    for (int r = 0; r < 4; ++r) {
      int i = r * 256 + tid;           // 0..1023 float4s
      int krow  = i >> 5;              // 0..31
      int ncol4 = (i & 31) << 2;       // 0..124
      const float4* bp =
          (const float4*)(B + (long long)(k0 + krow) * ldb + (n0 + ncol4));
      float4 v = *bp;
      sBt[ncol4 + 0][krow] = (_Float16)v.x;
      sBt[ncol4 + 1][krow] = (_Float16)v.y;
      sBt[ncol4 + 2][krow] = (_Float16)v.z;
      sBt[ncol4 + 3][krow] = (_Float16)v.w;
    }
    __syncthreads();

    // ---- A fragment: row m, K chunks [8*lhi..+7] and [16+8*lhi..+7] ----
    const int mrow = wm * 16 + lm;
    v8h alo = *(const v8h*)&sA[mrow][8 * lhi];
    v8h ahi = *(const v8h*)&sA[mrow][16 + 8 * lhi];
    v16h af = __builtin_shufflevector(alo, ahi,
        0,1,2,3,4,5,6,7, 8,9,10,11,12,13,14,15);

    // ---- B fragments: col n, K chunks [8*lhi..+7] and [16+8*lhi..+7] ----
    #pragma unroll
    for (int s = 0; s < 2; ++s) {
      int ncol = wn * 32 + s * 16 + lm;
      v8h blo = *(const v8h*)&sBt[ncol][8 * lhi];
      v8h bhi = *(const v8h*)&sBt[ncol][16 + 8 * lhi];
      v16h bf = __builtin_shufflevector(blo, bhi,
          0,1,2,3,4,5,6,7, 8,9,10,11,12,13,14,15);
      if (s == 0)
        c0 = __builtin_amdgcn_wmma_f32_16x16x32_f16(
            false, af, false, bf, (short)0, c0, false, false);
      else
        c1 = __builtin_amdgcn_wmma_f32_16x16x32_f16(
            false, af, false, bf, (short)0, c1, false, false);
    }
    __syncthreads();
  }

  // ---- epilogue: D row m = vv + 8*lhi, col n = lane&15 within tile ----
  #pragma unroll
  for (int s = 0; s < 2; ++s) {
    const v8f* cs = s ? &c1 : &c0;
    const int n  = n0 + wn * 32 + s * 16 + lm;   // always < N (N%128==0)
    const float bb = bias[n];                     // hoisted, unconditional
    const int mbase = m0 + wm * 16 + 8 * lhi;
    if (mbase + 8 <= M) {
      // fast path: full 8-row strip in range -> unconditional stores,
      // incremental addressing (no per-store 64-bit multiply / exec churn)
      if (FLAGS & 2) {
        _Float16* p = (_Float16*)Cout + (long long)mbase * ldc + n;
        #pragma unroll
        for (int vv = 0; vv < 8; ++vv) {
          float val = (*cs)[vv] + bb;
          if (FLAGS & 1) val = fmaxf(val, 0.0f);
          *p = (_Float16)val;
          p += ldc;
        }
      } else {
        float* p = (float*)Cout + (long long)mbase * ldc + n;
        #pragma unroll
        for (int vv = 0; vv < 8; ++vv) {
          float val = (*cs)[vv] + bb;
          if (FLAGS & 1) val = fmaxf(val, 0.0f);
          *p = val;
          p += ldc;
        }
      }
    } else {
      // edge path: only the last M-block of the value GEMM lands here
      #pragma unroll
      for (int vv = 0; vv < 8; ++vv) {
        int m = mbase + vv;
        if (m < M) {
          float val = (*cs)[vv] + bb;
          if (FLAGS & 1) val = fmaxf(val, 0.0f);
          if (FLAGS & 2)
            ((_Float16*)Cout)[(long long)m * ldc + n] = (_Float16)val;
          else
            ((float*)Cout)[(long long)m * ldc + n] = val;
        }
      }
    }
  }
}

// ---------------------------------------------------------------------------
// Flash-style self attention. Block = 128 threads, one query per thread.
// grid.x = B*NH*3 (3 query tiles of 128). K/V tiles of 100 keys staged in LDS.
// qk: [ROWS][512] (q cols 0..255, k cols 256..511), vbuf: [ROWS][256].
// ---------------------------------------------------------------------------
#define KT 100
__global__ __launch_bounds__(128) void attn_flash(
    const float* __restrict__ qk, const float* __restrict__ vbuf,
    float* __restrict__ out)
{
  __shared__ float Ks[KT][32];
  __shared__ float Vs[KT][32];
  int bid = blockIdx.x;
  int qt = bid % 3;
  int h  = (bid / 3) % NHD;
  int b  = bid / (3 * NHD);
  int tid = threadIdx.x;
  int qi = qt * 128 + tid;
  bool act = qi < NQ;

  const float scale = 0.17677669529663687f; // 32^-0.5
  float qreg[32], acc[32];
  float mval = -1e30f, lsum = 0.0f;
  if (act) {
    const float* qp = qk + (long long)(b * NQ + qi) * 512 + h * 32;
    #pragma unroll
    for (int d = 0; d < 32; ++d) { qreg[d] = qp[d] * scale; acc[d] = 0.0f; }
  }

  for (int t = 0; t < 3; ++t) {
    // 3200 floats per tile, as float4: 800 per array, 128 threads
    for (int e = tid; e < KT * 8; e += 128) {
      int kr = e >> 3, d4 = (e & 7) << 2;
      int kq = t * KT + kr;
      *(float4*)&Ks[kr][d4] =
          *(const float4*)(qk + (long long)(b * NQ + kq) * 512 + 256 + h * 32 + d4);
      *(float4*)&Vs[kr][d4] =
          *(const float4*)(vbuf + (long long)(b * NQ + kq) * 256 + h * 32 + d4);
    }
    __syncthreads();
    if (act) {
      for (int kr = 0; kr < KT; ++kr) {
        float s = 0.0f;
        #pragma unroll
        for (int d4 = 0; d4 < 8; ++d4) {
          float4 kv = *(const float4*)&Ks[kr][d4 << 2];
          s += qreg[d4*4+0]*kv.x + qreg[d4*4+1]*kv.y
             + qreg[d4*4+2]*kv.z + qreg[d4*4+3]*kv.w;
        }
        float mn = fmaxf(mval, s);
        float corr = __expf(mval - mn);
        float p    = __expf(s - mn);
        lsum = lsum * corr + p;
        #pragma unroll
        for (int d4 = 0; d4 < 8; ++d4) {
          float4 vv = *(const float4*)&Vs[kr][d4 << 2];
          acc[d4*4+0] = acc[d4*4+0]*corr + p*vv.x;
          acc[d4*4+1] = acc[d4*4+1]*corr + p*vv.y;
          acc[d4*4+2] = acc[d4*4+2]*corr + p*vv.z;
          acc[d4*4+3] = acc[d4*4+3]*corr + p*vv.w;
        }
        mval = mn;
      }
    }
    __syncthreads();
  }
  if (act) {
    float inv = 1.0f / lsum;
    float* op = out + (long long)(b * NQ + qi) * 256 + h * 32;
    #pragma unroll
    for (int d = 0; d < 32; ++d) op[d] = acc[d] * inv;
  }
}

// ---------------------------------------------------------------------------
// Deformable sampling. Block = 256 threads = 8 waves; block per (b,q),
// wave per head, lane per head-dim. Softmax over 16 (l,p) logits via wave ops.
// value stored fp16 (fits in 192MB L2) laid out [B*LIN][256] (col = h*32+d).
// ---------------------------------------------------------------------------
__global__ __launch_bounds__(256) void deform_sample(
    const float* __restrict__ offb,   // [ROWS][256]
    const float* __restrict__ awl,    // [ROWS][128]
    const _Float16* __restrict__ value,
    const float* __restrict__ ref,    // [ROWS][NL][2]
    float* __restrict__ out)          // [ROWS][256]
{
  const int starts[4] = {0, 10000, 12500, 13125};
  const int Hs[4] = {100, 50, 25, 13};
  const int Ws[4] = {100, 50, 25, 13};

  int row  = blockIdx.x;            // b*NQ + q
  int b    = row / NQ;
  int h    = threadIdx.x >> 5;
  int lane = threadIdx.x & 31;

  // softmax over the 16 (level, point) attention logits
  float logit = (lane < 16) ? awl[(long long)row * 128 + h * 16 + lane] : -1e30f;
  float m = warp_max(logit);
  float e = (lane < 16) ? __expf(logit - m) : 0.0f;
  float s = warp_sum(e);
  float p = e / s;

  const float* offp = offb + (long long)row * 256 + h * 32;
  float acc = 0.0f;

  for (int l = 0; l < NL; ++l) {
    int W = Ws[l], H = Hs[l], st = starts[l];
    float refx = ref[((long long)row * NL + l) * 2 + 0];
    float refy = ref[((long long)row * NL + l) * 2 + 1];
    #pragma unroll
    for (int pp = 0; pp < NP; ++pp) {
      int j = l * NP + pp;
      float wj = __shfl(p, j, 32);
      float ox = offp[j * 2 + 0];
      float oy = offp[j * 2 + 1];
      float x = (refx + ox / (float)W) * (float)W - 0.5f;
      float y = (refy + oy / (float)H) * (float)H - 0.5f;
      float x0 = floorf(x), y0 = floorf(y);
      #pragma unroll
      for (int dy = 0; dy < 2; ++dy) {
        #pragma unroll
        for (int dx = 0; dx < 2; ++dx) {
          float xi = x0 + (float)dx, yi = y0 + (float)dy;
          float wgt = (1.0f - fabsf(x - xi)) * (1.0f - fabsf(y - yi));
          if (xi >= 0.0f && xi < (float)W && yi >= 0.0f && yi < (float)H) {
            int xc = (int)xi, yc = (int)yi;
            long long idx =
                ((long long)b * LIN + st + yc * W + xc) * 256 + h * 32 + lane;
            acc += wj * wgt * (float)value[idx];
          }
        }
      }
    }
  }
  out[(long long)row * 256 + h * 32 + lane] = acc;
}

// ---------------------------------------------------------------------------
// out[row] = LayerNorm(res[row] + x[row]) * g + b   (C=256, wave per row)
// ---------------------------------------------------------------------------
__global__ __launch_bounds__(256) void add_ln(
    const float* __restrict__ res, const float* __restrict__ x,
    const float* __restrict__ g, const float* __restrict__ bta,
    float* __restrict__ out, int rows)
{
  int w = threadIdx.x >> 5, lane = threadIdx.x & 31;
  int row = blockIdx.x * 8 + w;
  if (row >= rows) return;
  const float* rp = res + (long long)row * 256;
  const float* xp = x   + (long long)row * 256;
  float v[8];
  float sum = 0.0f;
  #pragma unroll
  for (int i = 0; i < 8; ++i) {
    int c = i * 32 + lane;
    v[i] = rp[c] + xp[c];
    sum += v[i];
  }
  sum = warp_sum(sum);
  float mu = sum * (1.0f / 256.0f);
  float var = 0.0f;
  #pragma unroll
  for (int i = 0; i < 8; ++i) { float d = v[i] - mu; var += d * d; }
  var = warp_sum(var) * (1.0f / 256.0f);
  float inv = rsqrtf(var + 1e-5f);
  #pragma unroll
  for (int i = 0; i < 8; ++i) {
    int c = i * 32 + lane;
    out[(long long)row * 256 + c] = (v[i] - mu) * inv * g[c] + bta[c];
  }
}

__global__ __launch_bounds__(256) void add2(
    const float* __restrict__ a, const float* __restrict__ b,
    float* __restrict__ o, int n)
{
  int i = blockIdx.x * 256 + threadIdx.x;
  if (i < n) o[i] = a[i] + b[i];
}

// ---------------------------------------------------------------------------
extern "C" void kernel_launch(void* const* d_in, const int* in_sizes, int n_in,
                              void* d_out, int out_size, void* d_ws, size_t ws_size,
                              hipStream_t stream) {
  (void)in_sizes; (void)n_in; (void)out_size; (void)ws_size;
  // inputs in setup_inputs() order
  const float* tgt  = (const float*)d_in[0];
  const float* qpos = (const float*)d_in[1];
  const float* refp = (const float*)d_in[2];
  const float* src  = (const float*)d_in[3];
  // d_in[4] = spatial_shapes (int32, hardcoded)
  const float* Wv    = (const float*)d_in[5];
  const float* bv    = (const float*)d_in[6];
  const float* Woff  = (const float*)d_in[7];
  const float* boff  = (const float*)d_in[8];
  const float* Wa    = (const float*)d_in[9];
  const float* ba    = (const float*)d_in[10];
  const float* Wout  = (const float*)d_in[11];
  const float* bout  = (const float*)d_in[12];
  const float* Wqkv  = (const float*)d_in[13];
  const float* bqkv  = (const float*)d_in[14];
  const float* Wproj = (const float*)d_in[15];
  const float* bproj = (const float*)d_in[16];
  const float* ln1g  = (const float*)d_in[17];
  const float* ln1b  = (const float*)d_in[18];
  const float* ln2g  = (const float*)d_in[19];
  const float* ln2b  = (const float*)d_in[20];
  const float* ln3g  = (const float*)d_in[21];
  const float* ln3b  = (const float*)d_in[22];
  const float* W1    = (const float*)d_in[23];
  const float* b1    = (const float*)d_in[24];
  const float* W2    = (const float*)d_in[25];
  const float* b2    = (const float*)d_in[26];

  // workspace bump allocator (floats)
  float* wsp = (float*)d_ws;
  size_t off = 0;
  auto alloc = [&](size_t n) { float* p = wsp + off; off += n; return p; };
  float* qk_in = alloc((size_t)ROWS * CC);        // tgt + query_pos
  float* qkbuf = alloc((size_t)ROWS * 2 * CC);    // q|k
  float* vproj = alloc((size_t)ROWS * CC);        // v
  float* attnb = alloc((size_t)ROWS * CC);
  float* satmp = alloc((size_t)ROWS * CC);
  float* tgta  = alloc((size_t)ROWS * CC);
  float* qca   = alloc((size_t)ROWS * CC);
  float* offb  = alloc((size_t)ROWS * CC);
  float* awlb  = alloc((size_t)ROWS * 128);
  float* capre = alloc((size_t)ROWS * CC);
  float* cab   = alloc((size_t)ROWS * CC);
  float* tgtb  = alloc((size_t)ROWS * CC);
  float* ffh   = alloc((size_t)ROWS * DFF);
  float* ffo   = alloc((size_t)ROWS * CC);
  float* valH  = alloc(((size_t)SRCROWS * CC + 1) / 2); // fp16 value buffer

  auto g2 = [](int M, int N) { return dim3((N + 127) / 128, (M + 31) / 32); };
  const int n2400 = ROWS * CC;

  // ---- self-attention block ----
  add2<<<(n2400 + 255) / 256, 256, 0, stream>>>(tgt, qpos, qk_in, n2400);
  gemm_wmma_f16<0><<<g2(ROWS, 512), 256, 0, stream>>>(
      qk_in, CC, Wqkv, 3 * CC, bqkv, qkbuf, 512, ROWS, 512, CC);
  gemm_wmma_f16<0><<<g2(ROWS, CC), 256, 0, stream>>>(
      tgt, CC, Wqkv + 2 * CC, 3 * CC, bqkv + 2 * CC, vproj, CC, ROWS, CC, CC);
  attn_flash<<<BB * NHD * 3, 128, 0, stream>>>(qkbuf, vproj, attnb);
  gemm_wmma_f16<0><<<g2(ROWS, CC), 256, 0, stream>>>(
      attnb, CC, Wproj, CC, bproj, satmp, CC, ROWS, CC, CC);
  add_ln<<<(ROWS + 7) / 8, 256, 0, stream>>>(tgt, satmp, ln2g, ln2b, tgta, ROWS);

  // ---- deformable cross-attention block ----
  add2<<<(n2400 + 255) / 256, 256, 0, stream>>>(tgta, qpos, qca, n2400);
  gemm_wmma_f16<2><<<g2(SRCROWS, CC), 256, 0, stream>>>(      // value proj (big, f16 out)
      src, CC, Wv, CC, bv, valH, CC, SRCROWS, CC, CC);
  gemm_wmma_f16<0><<<g2(ROWS, CC), 256, 0, stream>>>(
      qca, CC, Woff, CC, boff, offb, CC, ROWS, CC, CC);
  gemm_wmma_f16<0><<<g2(ROWS, 128), 256, 0, stream>>>(
      qca, CC, Wa, 128, ba, awlb, 128, ROWS, 128, CC);
  deform_sample<<<ROWS, 256, 0, stream>>>(
      offb, awlb, (const _Float16*)valH, refp, capre);
  gemm_wmma_f16<0><<<g2(ROWS, CC), 256, 0, stream>>>(
      capre, CC, Wout, CC, bout, cab, CC, ROWS, CC, CC);
  add_ln<<<(ROWS + 7) / 8, 256, 0, stream>>>(tgta, cab, ln1g, ln1b, tgtb, ROWS);

  // ---- FFN ----
  gemm_wmma_f16<1><<<g2(ROWS, DFF), 256, 0, stream>>>(        // relu
      tgtb, CC, W1, DFF, b1, ffh, DFF, ROWS, DFF, CC);
  gemm_wmma_f16<0><<<g2(ROWS, CC), 256, 0, stream>>>(
      ffh, DFF, W2, CC, b2, ffo, CC, ROWS, CC, DFF);
  add_ln<<<(ROWS + 7) / 8, 256, 0, stream>>>(
      tgtb, ffo, ln3g, ln3b, (float*)d_out, ROWS);
}
